// GCN_79989470921103
// MI455X (gfx1250) — matrix operand order repaired and checked by reading, not compiled, not run on hardware
//
#include <hip/hip_runtime.h>
#include <hip/hip_bf16.h>

typedef __attribute__((ext_vector_type(16))) __bf16 v16bf;
typedef __attribute__((ext_vector_type(8)))  float  v8f;

// f32 -> bf16 bits (low 16), round-to-nearest-even, pure integer ops
__device__ __forceinline__ unsigned bfbits(float f) {
    union { float f; unsigned u; } v; v.f = f;
    return (v.u + 0x7FFFu + ((v.u >> 16) & 1u)) >> 16;
}
__device__ __forceinline__ unsigned bfpack2(float a, float b) {
    return bfbits(a) | (bfbits(b) << 16);
}

union Frag { uint4 q[2]; v16bf v; };

// ---------------------------------------------------------------------------
// Y[r, c] = act( sum_k X[r, k] * W[c, k] + bias[c] )   (W row-major [64 x K])
// Wave32 WMMA bf16 GEMM, LDS-staged operands, K compile-time.
// Block = 256 thr = 8 waves -> 32 rows x 64 cols. nrows % 32 == 0.
// ---------------------------------------------------------------------------
template<int K>
__global__ __launch_bounds__(256)
void gemm_nt_wmma(const float* __restrict__ X, int ldX,
                  const float* __restrict__ W,
                  const float* __restrict__ bias,
                  float* __restrict__ Y, int ldY,
                  int relu)
{
    constexpr int KP = K + 8;                 // padded stride (keeps 16B align)
    __shared__ __bf16 Ash[32][KP];
    __shared__ __bf16 Bsh[64][KP];

    const int tid    = threadIdx.x;
    const int rowblk = blockIdx.x * 32;

    // ---- stage A panel (32 x K) as bf16, float4 -> packed uint2 stores ----
    constexpr int KQ = K / 4;
#pragma unroll 2
    for (int idx = tid; idx < 32 * KQ; idx += 256) {
        int row = idx / KQ, kq = idx % KQ;
        const float4 f = *(const float4*)(X + (size_t)(rowblk + row) * (size_t)ldX + kq * 4);
        *(uint2*)&Ash[row][kq * 4] = make_uint2(bfpack2(f.x, f.y), bfpack2(f.z, f.w));
    }
    // ---- stage B = W (64 x K) as bf16 ----
#pragma unroll 2
    for (int idx = tid; idx < 64 * KQ; idx += 256) {
        int row = idx / KQ, kq = idx % KQ;
        const float4 f = *(const float4*)(W + (size_t)row * K + kq * 4);
        *(uint2*)&Bsh[row][kq * 4] = make_uint2(bfpack2(f.x, f.y), bfpack2(f.z, f.w));
    }
    __syncthreads();

    const int wave = tid >> 5;
    const int lane = tid & 31;
    const int l15  = lane & 15;
    const int hi   = lane >> 4;               // lane half
    const int m    = (wave >> 2) * 16 + l15;  // A row within panel
    const int n    = (wave & 3) * 16 + l15;   // output column
    const int ah   = hi * 8;                  // A-fragment lane-half K offset
    const int bh   = hi * 8;                  // B-fragment lane-half K offset

    v8f acc = {0.f, 0.f, 0.f, 0.f, 0.f, 0.f, 0.f, 0.f};

#pragma unroll
    for (int k0 = 0; k0 < K; k0 += 32) {
        Frag a, b;
        a.q[0] = *(const uint4*)&Ash[m][k0 + ah];        // K = k0+ah    .. +7
        a.q[1] = *(const uint4*)&Ash[m][k0 + ah + 16];   // K = k0+ah+16 .. +23
        b.q[0] = *(const uint4*)&Bsh[n][k0 + bh];        // K = k0+bh    .. +7
        b.q[1] = *(const uint4*)&Bsh[n][k0 + bh + 16];   // K = k0+bh+16 .. +23
        acc = __builtin_amdgcn_wmma_f32_16x16x32_bf16(
                  false, a.v, false, b.v, (short)0, acc, false, false);
    }

    const float bb   = bias ? bias[n] : 0.f;
    const int   rtop = rowblk + (wave >> 2) * 16 + hi * 8;
#pragma unroll
    for (int r = 0; r < 8; ++r) {
        float v = acc[r] + bb;
        if (relu) v = fmaxf(v, 0.f);
        Y[(size_t)(rtop + r) * (size_t)ldY + n] = v;
    }
}

// ---------------------------------------------------------------------------
__global__ void fill_zero(float* __restrict__ p, long long n) {
    long long i = (long long)blockIdx.x * blockDim.x + threadIdx.x;
    if (i < n) p[i] = 0.f;
}

__global__ void deg_count(const int* __restrict__ dst, float* __restrict__ deg, int e) {
    int i = blockIdx.x * blockDim.x + threadIdx.x;
    if (i < e) atomicAdd(&deg[dst[i]], 1.0f);
}

__global__ void make_dis(float* __restrict__ deg, int n) {
    int i = blockIdx.x * blockDim.x + threadIdx.x;
    if (i < n) deg[i] = rsqrtf(deg[i] + 1.0f);
}

// agg[i,f] = hlin[i,f] * dis[i]^2 + bias[f]   (self-loop term + conv bias)
__global__ void self_init(const float* __restrict__ hlin, const float* __restrict__ dis,
                          const float* __restrict__ bias, float* __restrict__ agg,
                          long long total)
{
    long long i = (long long)blockIdx.x * blockDim.x + threadIdx.x;
    if (i >= total) return;
    int node = (int)(i >> 6);
    int f    = (int)(i & 63);
    float d  = dis[node];
    agg[i] = hlin[i] * d * d + bias[f];
}

// agg[dst] += hlin[src] * dis[src]*dis[dst]; one wave per edge, float2/lane
__global__ void edge_agg(const int* __restrict__ src, const int* __restrict__ dst,
                         const float* __restrict__ dis, const float* __restrict__ hlin,
                         float* __restrict__ agg, long long nthread)
{
    long long i = (long long)blockIdx.x * blockDim.x + threadIdx.x;
    if (i >= nthread) return;
    long long e = i >> 5;
    int lane = (int)(i & 31);
    int s = src[e], d = dst[e];
    float c = dis[s] * dis[d];
    const float2* hp = (const float2*)(hlin + (size_t)s * 64);
    float2 h = hp[lane];
    float* ap = agg + (size_t)d * 64 + (size_t)lane * 2;
    atomicAdd(ap,     h.x * c);
    atomicAdd(ap + 1, h.y * c);
}

// pooled[batch[node], f] = max(...)  via int atomicMax (values >= 0 post-ReLU)
__global__ void pool_max(const float* __restrict__ hcat, const int* __restrict__ batch,
                         float* __restrict__ pooled, long long total)
{
    long long i = (long long)blockIdx.x * blockDim.x + threadIdx.x;
    if (i >= total) return;
    int node = (int)(i >> 8);
    int f    = (int)(i & 255);
    int g = batch[node];
    float v = hcat[i];
    atomicMax((int*)&pooled[(size_t)g * 256 + f], __float_as_int(v));
}

// out[g,t] = sum_d hid[g,d] * W2[t,d] + b2[t]
__global__ void head_kernel(const float* __restrict__ hid, const float* __restrict__ W2,
                            const float* __restrict__ b2, float* __restrict__ out,
                            int G, int T)
{
    int i = blockIdx.x * blockDim.x + threadIdx.x;
    if (i >= G * T) return;
    int g = i / T, t = i % T;
    const float* h = hid + (size_t)g * 64;
    const float* w = W2  + (size_t)t * 64;
    float s = b2[t];
#pragma unroll 8
    for (int d = 0; d < 64; ++d) s += h[d] * w[d];
    out[i] = s;
}

// ---------------------------------------------------------------------------
extern "C" void kernel_launch(void* const* d_in, const int* in_sizes, int n_in,
                              void* d_out, int out_size, void* d_ws, size_t ws_size,
                              hipStream_t stream)
{
    const int F_IN = 128, D = 64, T = 10;
    const int N = in_sizes[0] / F_IN;
    const int E = in_sizes[1] / 2;
    const int G = out_size / T;

    const float* x     = (const float*)d_in[0];
    const int*   ei    = (const int*)d_in[1];
    const int*   batch = (const int*)d_in[2];
    const float* Wc[4] = {(const float*)d_in[4], (const float*)d_in[6],
                          (const float*)d_in[8], (const float*)d_in[10]};
    const float* bc[4] = {(const float*)d_in[5], (const float*)d_in[7],
                          (const float*)d_in[9], (const float*)d_in[11]};
    const float* Wmax = (const float*)d_in[12];
    const float* bmax = (const float*)d_in[13];
    const float* W1   = (const float*)d_in[14];
    const float* b1   = (const float*)d_in[15];
    const float* W2   = (const float*)d_in[16];
    const float* b2   = (const float*)d_in[17];
    float* out = (float*)d_out;

    const int* src = ei;
    const int* dst = ei + E;

    // workspace carve-up (floats), 64-float aligned
    float* ws = (float*)d_ws;
    size_t off = 0;
    auto carve = [&](size_t n) { float* p = ws + off; off += (n + 63) & ~(size_t)63; return p; };
    float* dis    = carve((size_t)N);            // degree, then rsqrt(deg+1)
    float* hlin   = carve((size_t)N * D);        // X @ Wc^T
    float* agg    = carve((size_t)N * D);        // normalized aggregation
    float* hcat   = carve((size_t)N * 4 * D);    // concat of 4 layer outputs
    float* pooled = carve((size_t)G * 4 * D);    // segment-max
    float* hid    = carve((size_t)G * D);        // MLP hidden
    (void)ws_size;

    const int BT = 256;
    auto blks = [&](long long n) { return (unsigned)((n + BT - 1) / BT); };

    // degree -> dis
    fill_zero<<<blks(N), BT, 0, stream>>>(dis, N);
    deg_count<<<blks(E), BT, 0, stream>>>(dst, dis, E);
    make_dis<<<blks(N), BT, 0, stream>>>(dis, N);

    for (int l = 0; l < 4; ++l) {
        // hlin = Xin @ Wc[l]^T (no bias yet)
        if (l == 0) {
            gemm_nt_wmma<128><<<N / 32, BT, 0, stream>>>(x, F_IN, Wc[0], nullptr, hlin, D, 0);
        } else {
            gemm_nt_wmma<64><<<N / 32, BT, 0, stream>>>(hcat + (size_t)(l - 1) * D, 4 * D,
                                                        Wc[l], nullptr, hlin, D, 0);
        }
        // agg = self-loop term + bias
        self_init<<<blks((long long)N * D), BT, 0, stream>>>(hlin, dis, bc[l], agg,
                                                             (long long)N * D);
        // agg[dst] += hlin[src] * dis[src]*dis[dst]
        edge_agg<<<blks((long long)E * 32), BT, 0, stream>>>(src, dst, dis, hlin, agg,
                                                             (long long)E * 32);
        // hcat[:, 64l:64l+64] = relu(agg @ Wmax^T + bmax)
        gemm_nt_wmma<64><<<N / 32, BT, 0, stream>>>(agg, D, Wmax, bmax,
                                                    hcat + (size_t)l * D, 4 * D, 1);
    }

    // segment-max pooling (post-ReLU values >= 0, int-bit max is monotonic)
    fill_zero<<<blks((long long)G * 4 * D), BT, 0, stream>>>(pooled, (long long)G * 4 * D);
    pool_max<<<blks((long long)N * 4 * D), BT, 0, stream>>>(hcat, batch, pooled,
                                                            (long long)N * 4 * D);

    // hid = relu(pooled @ W1^T + b1)   (G=512 rows, K=256, 64 cols -> WMMA path)
    gemm_nt_wmma<256><<<G / 32, BT, 0, stream>>>(pooled, 4 * D, W1, b1, hid, D, 1);

    // out = hid @ W2^T + b2
    head_kernel<<<blks((long long)G * T), BT, 0, stream>>>(hid, W2, b2, out, G, T);
}
